// Discrinator_28020366639587
// MI455X (gfx1250) — matrix operand order repaired and checked by reading, not compiled, not run on hardware
//
#include <hip/hip_runtime.h>

typedef __attribute__((ext_vector_type(16))) _Float16 v16h;
typedef __attribute__((ext_vector_type(8)))  _Float16 v8h;
typedef __attribute__((ext_vector_type(8)))  float    v8f;

// ---------------------------------------------------------------------------
// decompose_image: x[32,3,256,256] f32 NCHW -> dec[32,256,256,32] f16 NHWC
// channels: 0..2 = box8, 3..5 = (box2-box8)*15, 6..8 = (x-box2)*30, 9..31 = 0
// box blur uses BORDER_REFLECT_101; k=8 window = [-4,+3], k=2 window = [-1,0].
// ---------------------------------------------------------------------------
__global__ void __launch_bounds__(256) decompose_kernel(
    const float* __restrict__ x, _Float16* __restrict__ dec)
{
  int idx = blockIdx.x * 256 + threadIdx.x;        // b*65536 + y*256 + x
  int xq = idx & 255, yq = (idx >> 8) & 255, b = idx >> 16;
  const float* xb = x + (size_t)b * 3 * 65536;
  _Float16* o = dec + (size_t)idx * 32;
  _Float16 tmp[32];
#pragma unroll
  for (int c = 9; c < 32; ++c) tmp[c] = (_Float16)0.f;
#pragma unroll
  for (int c = 0; c < 3; ++c) {
    const float* p = xb + c * 65536;
    float s8 = 0.f;
#pragma unroll
    for (int dy = -4; dy <= 3; ++dy) {
      int yy = yq + dy; yy = yy < 0 ? -yy : (yy > 255 ? 510 - yy : yy);
      const float* row = p + yy * 256;
#pragma unroll
      for (int dx = -4; dx <= 3; ++dx) {
        int xx = xq + dx; xx = xx < 0 ? -xx : (xx > 255 ? 510 - xx : xx);
        s8 += row[xx];
      }
    }
    s8 *= (1.f / 64.f);
    int ym = yq - 1; ym = ym < 0 ? 1 : ym;   // reflect-101
    int xm = xq - 1; xm = xm < 0 ? 1 : xm;
    float s2 = 0.25f * (p[ym * 256 + xm] + p[ym * 256 + xq] +
                        p[yq * 256 + xm] + p[yq * 256 + xq]);
    float ctr = p[yq * 256 + xq];
    tmp[c]     = (_Float16)s8;
    tmp[c + 3] = (_Float16)((s2 - s8) * 15.f);
    tmp[c + 6] = (_Float16)((ctr - s2) * 30.f);
  }
  v8h* ov = (v8h*)o;
#pragma unroll
  for (int i = 0; i < 4; ++i) ov[i] = *(const v8h*)(tmp + 8 * i);
}

// ---------------------------------------------------------------------------
// Zero page (redirect target for out-of-bounds conv taps)
// ---------------------------------------------------------------------------
__global__ void zero_page_kernel(unsigned int* zp)
{
  zp[threadIdx.x] = 0u;   // 64 threads * 4B = 256B
}

// ---------------------------------------------------------------------------
// Weight prep (BN layers): OIHW f32 -> wt[kc][t][co][32] f16 (K-chunk-major,
// so each 32-wide K chunk over all 9 taps is one contiguous LDS-stageable
// block), plus folded BN scale/shift f32.
// ---------------------------------------------------------------------------
__global__ void __launch_bounds__(256) prep_layer(
    const float* __restrict__ w, const float* __restrict__ gamma,
    const float* __restrict__ beta, const float* __restrict__ mean,
    const float* __restrict__ var,
    _Float16* __restrict__ wt, float* __restrict__ scale,
    float* __restrict__ shift, int Cout, int Cin, int CinP)
{
  int i = blockIdx.x * 256 + threadIdx.x;
  int total = 9 * Cout * CinP;
  if (i < total) {
    int ci32 = i & 31;
    int co   = (i >> 5) % Cout;
    int t    = (i / (32 * Cout)) % 9;
    int kc   = i / (32 * Cout * 9);
    int ci   = kc * 32 + ci32;
    float v  = (ci < Cin) ? w[(size_t)(co * Cin + ci) * 9 + t] : 0.f;
    wt[i] = (_Float16)v;
  }
  if (i < Cout) {
    float inv = rsqrtf(var[i] + 1e-5f);
    float sc = gamma[i] * inv;
    scale[i] = sc;
    shift[i] = beta[i] - mean[i] * sc;
  }
}

// Heads: fuse b4l (co 0..3) and b4s (co 4..7) into CoutP=16, zero-padded.
__global__ void __launch_bounds__(256) prep_head(
    const float* __restrict__ wl, const float* __restrict__ bl,
    const float* __restrict__ wsrc, const float* __restrict__ bsrc,
    _Float16* __restrict__ wt, float* __restrict__ scale,
    float* __restrict__ shift, int Cin)
{
  int i = blockIdx.x * 256 + threadIdx.x;
  int total = 9 * 16 * Cin;
  if (i < total) {
    int ci32 = i & 31;
    int co   = (i >> 5) % 16;
    int t    = (i / (32 * 16)) % 9;
    int kc   = i / (32 * 16 * 9);
    int ci   = kc * 32 + ci32;
    float v = 0.f;
    if (co < 4)      v = wl[(size_t)(co * Cin + ci) * 9 + t];
    else if (co < 8) v = wsrc[(size_t)((co - 4) * Cin + ci) * 9 + t];
    wt[i] = (_Float16)v;
  }
  if (i < 16) {
    scale[i] = 1.f;
    shift[i] = (i < 4) ? bl[i] : ((i < 8) ? bsrc[i - 4] : 0.f);
  }
}

// ---------------------------------------------------------------------------
// Implicit-GEMM 3x3 conv via v_wmma_f32_16x16x32_f16.
//   act: NHWC f16 (C = CinP, multiple of 32); wt: [kc][t][co][32] f16.
//   M = b*Ho*Wo flattened. WG = 256 threads = 8 waves:
//   WM m-subtiles x (8/WM) n-groups, NT n-tiles (of 16) per wave.
//   Out-of-bounds taps read a global 256B zero page (pointer select, no
//   per-element masking). One LDS stage + barrier pair per 32-wide K chunk.
// Fragment layouts used here (contiguous K, same order for A and B):
//   A 16x32 f16 : lane holds row m=lane&15; halves 0-15 = K kb..kb+15, kb = 16*(lane>=16).
//   B 32x16 f16 : lane holds col n=lane&15, K slice 16*(lane>=16)..+16.
//   C/D 16x16 f32: VGPR v -> (m = v + 8*(lane>=16), n = lane&15).
// ---------------------------------------------------------------------------
template<int NT, int WM, bool HEADS>
__global__ void __launch_bounds__(256) conv3x3_wmma(
    const _Float16* __restrict__ act, const _Float16* __restrict__ wt,
    const float* __restrict__ scale, const float* __restrict__ shift,
    _Float16* __restrict__ outh, float* __restrict__ outf,
    const _Float16* __restrict__ zpage,
    int B, int Hin, int Win, int CinP, int Ho, int Wo, int CoutP,
    int stride, int relu, int kcN)
{
  extern __shared__ _Float16 lw[];
  const int tid = threadIdx.x;
  const int wid = tid >> 5, lane = tid & 31;
  const int laneHi = lane >> 4, lane15 = lane & 15;
  const int mt = wid % WM, nw = wid / WM;
  const int n0 = nw * NT * 16;
  const int HoWo = Ho * Wo;

  long mrow = (long)blockIdx.x * (16 * WM) + mt * 16 + lane15;
  int bA  = (int)(mrow / HoWo);
  int rA  = (int)(mrow % HoWo);
  int oyA = rA / Wo, oxA = rA - (rA / Wo) * Wo;

  // Per-tap A base pointers; OOB taps redirected to the zero page once here.
  const _Float16* ab[9];
#pragma unroll
  for (int t = 0; t < 9; ++t) {
    int ty = t / 3, tx = t - ty * 3;
    int iy = oyA * stride + ty - 1;
    int ix = oxA * stride + tx - 1;
    bool valid = (iy >= 0) & (iy < Hin) & (ix >= 0) & (ix < Win);
    const _Float16* p = act + (((size_t)bA * Hin + iy) * Win + ix) * CinP;
    ab[t] = valid ? p : zpage;
  }

  v8f acc[NT];
#pragma unroll
  for (int i = 0; i < NT; ++i) acc[i] = (v8f){};

  const int ldsHalves = 9 * CoutP * 32;      // one K-chunk for all taps
  const int ldsWords  = ldsHalves >> 1;
  unsigned int* lwW = (unsigned int*)lw;

  for (int kc = 0; kc < kcN; ++kc) {
    __syncthreads();
    const unsigned int* src = (const unsigned int*)(wt + (size_t)kc * ldsHalves);
    for (int i = tid; i < ldsWords; i += 256) lwW[i] = src[i];
    __syncthreads();

    const int kb = kc * 32 + laneHi * 16;
#pragma unroll
    for (int t = 0; t < 9; ++t) {
      v8h c1 = *(const v8h*)(ab[t] + kb);
      v8h c2 = *(const v8h*)(ab[t] + kb + 8);
      v16h a = __builtin_shufflevector(c1, c2, 0, 1, 2, 3, 4, 5, 6, 7,
                                       8, 9, 10, 11, 12, 13, 14, 15);
#pragma unroll
      for (int nt = 0; nt < NT; ++nt) {
        const _Float16* bp =
            lw + (((t * CoutP + n0 + nt * 16 + lane15) << 5) + laneHi * 16);
        v16h bf = *(const v16h*)bp;
        acc[nt] = __builtin_amdgcn_wmma_f32_16x16x32_f16(
            false, a, false, bf, (short)0, acc[nt], false, false);
      }
    }
  }

  // Epilogue: y = acc*scale + shift (+ReLU) -> NHWC f16, or heads -> NCHW f32.
  long rowbase = (long)blockIdx.x * (16 * WM) + mt * 16 + laneHi * 8;
#pragma unroll
  for (int nt = 0; nt < NT; ++nt) {
    int n = n0 + nt * 16 + lane15;
    float sc = scale[n], sh = shift[n];
#pragma unroll
    for (int v = 0; v < 8; ++v) {
      long row = rowbase + v;
      int b  = (int)(row / HoWo);
      int rr = (int)(row % HoWo);
      int oy = rr / Wo, ox = rr - (rr / Wo) * Wo;
      float val = acc[nt][v] * sc + sh;
      if (relu) val = fmaxf(val, 0.f);
      if (HEADS) {
        if (n < 8) {   // [out_l ; out_s], each NCHW [B,4,Ho,Wo] f32
          size_t half = (size_t)(n >> 2) * ((size_t)B * 4 * HoWo);
          outf[half + (((size_t)b * 4 + (n & 3)) * Ho + oy) * Wo + ox] = val;
        }
      } else {
        outh[(((size_t)b * Ho + oy) * Wo + ox) * CoutP + n] = (_Float16)val;
      }
    }
  }
}

// ---------------------------------------------------------------------------
// Host side
// ---------------------------------------------------------------------------
extern "C" void kernel_launch(void* const* d_in, const int* in_sizes, int n_in,
                              void* d_out, int out_size, void* d_ws, size_t ws_size,
                              hipStream_t stream)
{
  (void)in_sizes; (void)n_in; (void)out_size; (void)ws_size;
  const float* x = (const float*)d_in[0];

  struct LDef { int cin, cinP, cout; };
  const LDef L[8] = {{9, 32, 32}, {32, 32, 32}, {32, 32, 64}, {64, 64, 64},
                     {64, 64, 96}, {96, 96, 96}, {96, 96, 96}, {96, 96, 96}};

  char* ws = (char*)d_ws;
  size_t off = 0;
  auto alloc = [&](size_t n) {
    size_t o = off;
    off += (n + 255) & ~(size_t)255;
    return o;
  };

  // Ping-pong NHWC f16 activation buffers (each sized for 32ch @ 256^2)
  size_t buf0 = alloc((size_t)32 * 256 * 256 * 32 * 2);   // 128 MB
  size_t buf1 = alloc((size_t)32 * 256 * 256 * 32 * 2);   // 128 MB
  size_t buf2 = alloc((size_t)32 * 256 * 256 * 32 * 2);   // 128 MB
  size_t zpOff = alloc(256);                              // zero page

  size_t wtOff[9], scOff[9], shOff[9];
  for (int l = 0; l < 8; ++l) {
    wtOff[l] = alloc((size_t)9 * L[l].cout * L[l].cinP * 2);
    scOff[l] = alloc((size_t)L[l].cout * 4);
    shOff[l] = alloc((size_t)L[l].cout * 4);
  }
  wtOff[8] = alloc((size_t)9 * 16 * 96 * 2);
  scOff[8] = alloc((size_t)16 * 4);
  shOff[8] = alloc((size_t)16 * 4);

  zero_page_kernel<<<1, 64, 0, stream>>>((unsigned int*)(ws + zpOff));

  // --- weight / BN prep -----------------------------------------------------
  int pi = 1;
  for (int l = 0; l < 8; ++l) {
    const float* w  = (const float*)d_in[pi + 0];
    const float* g  = (const float*)d_in[pi + 1];
    const float* be = (const float*)d_in[pi + 2];
    const float* me = (const float*)d_in[pi + 3];
    const float* va = (const float*)d_in[pi + 4];
    pi += 5;
    int total = 9 * L[l].cout * L[l].cinP;
    prep_layer<<<(total + 255) / 256, 256, 0, stream>>>(
        w, g, be, me, va,
        (_Float16*)(ws + wtOff[l]), (float*)(ws + scOff[l]),
        (float*)(ws + shOff[l]), L[l].cout, L[l].cin, L[l].cinP);
  }
  {
    const float* wl = (const float*)d_in[pi + 0];
    const float* bl = (const float*)d_in[pi + 1];
    const float* wsrc = (const float*)d_in[pi + 2];
    const float* bsrc = (const float*)d_in[pi + 3];
    int total = 9 * 16 * 96;
    prep_head<<<(total + 255) / 256, 256, 0, stream>>>(
        wl, bl, wsrc, bsrc,
        (_Float16*)(ws + wtOff[8]), (float*)(ws + scOff[8]),
        (float*)(ws + shOff[8]), 96);
  }

  // --- decompose ------------------------------------------------------------
  decompose_kernel<<<(32 * 256 * 256) / 256, 256, 0, stream>>>(
      x, (_Float16*)(ws + buf0));

  // --- conv pipeline --------------------------------------------------------
  const int B = 32;
  const _Float16* zp = (const _Float16*)(ws + zpOff);

#define CONV(NT, WM, HEADS, inOff, outOff, li, Hin, CinP, Ho, CoutP, stride, relu, outfp) \
  conv3x3_wmma<NT, WM, HEADS><<<(B * (Ho) * (Ho)) / (16 * (WM)), 256,          \
                                (size_t)9 * (CoutP) * 32 * 2, stream>>>(        \
      (const _Float16*)(ws + (inOff)), (const _Float16*)(ws + wtOff[li]),       \
      (const float*)(ws + scOff[li]), (const float*)(ws + shOff[li]),           \
      (_Float16*)(ws + (outOff)), (outfp), zp,                                  \
      B, (Hin), (Hin), (CinP), (Ho), (Ho), (CoutP), (stride), (relu),           \
      (CinP) / 32)

  CONV(1, 4, false, buf0, buf1, 0, 256, 32, 256, 32, 1, 1, nullptr);  // stem
  CONV(1, 4, false, buf1, buf2, 1, 256, 32, 256, 32, 1, 1, nullptr);  // b1a
  CONV(2, 4, false, buf2, buf0, 2, 256, 32, 128, 64, 2, 1, nullptr);  // b1d
  CONV(2, 4, false, buf0, buf1, 3, 128, 64, 128, 64, 1, 1, nullptr);  // b2a
  CONV(3, 4, false, buf1, buf2, 4, 128, 64,  64, 96, 2, 1, nullptr);  // b2d
  CONV(3, 4, false, buf2, buf0, 5,  64, 96,  64, 96, 1, 1, nullptr);  // b3a
  CONV(3, 4, false, buf0, buf1, 6,  64, 96,  32, 96, 2, 1, nullptr);  // b3d
  CONV(3, 4, false, buf1, buf2, 7,  32, 96,  32, 96, 1, 1, nullptr);  // b4
  CONV(1, 8, true,  buf2, buf2, 8,  32, 96,  32, 16, 1, 0, (float*)d_out); // heads
#undef CONV
}
